// MultiHeadAttention_34368328303222
// MI455X (gfx1250) — compile-verified, matrix operand order checked by reading, not run on hardware
//
#include <hip/hip_runtime.h>
#include <hip/hip_bf16.h>

// ---------------------------------------------------------------------------
// MI455X (gfx1250) fused MHA:  projections -> attention -> out-proj + LN
// wave32, V_WMMA_F32_16X16X32_BF16 everywhere, fp32 accumulate.
// All WMMA operands are 16-byte vector loads (2 x b128 per fragment);
// the 536 MB attn stream is written with b128 non-temporal stores.
// ---------------------------------------------------------------------------

typedef __bf16 bf16_t;
typedef __attribute__((ext_vector_type(16))) __bf16 v16bf;
typedef __attribute__((ext_vector_type(8)))  __bf16 v8bf;
typedef __attribute__((ext_vector_type(8)))  float  v8f;
typedef __attribute__((ext_vector_type(4)))  float  v4f;

#define BATCH   8
#define SEQ     2048
#define DMODEL  128
#define NH      4
#define DK      32
#define ROWS    (BATCH * SEQ)      // 16384 token rows
#define LN_EPS  1e-5f

#define WMMA_BF16(A, B, C) \
  __builtin_amdgcn_wmma_f32_16x16x32_bf16(false, (A), false, (B), (short)0, (C), false, false)

static __device__ __forceinline__ v16bf combine8(v8bf lo, v8bf hi) {
  return __builtin_shufflevector(lo, hi, 0, 1, 2, 3, 4, 5, 6, 7,
                                         8, 9, 10, 11, 12, 13, 14, 15);
}

// A/B fragment from a row-contiguous bf16 matrix.
// Lane L supplies row M (A) / column N (B via pre-transposed storage) = base + L%16.
// Per ISA 7.12.2 the lane's 16 elements are K = hi+[0..7] and hi+16+[0..7]
// (hi = (L>=16)?8:0) -> exactly two contiguous 16-byte chunks.
static __device__ __forceinline__ v16bf frag_rows_bf16(const bf16_t* __restrict__ p,
                                                       int ld, int rowbase, int colbase, int lane) {
  const int m  = lane & 15;
  const int hi = (lane & 16) ? 8 : 0;
  const bf16_t* row = p + (size_t)(rowbase + m) * ld + colbase + hi;
  const v8bf lo = *(const v8bf*)(row);
  const v8bf h8 = *(const v8bf*)(row + 16);
  return combine8(lo, h8);
}

// Same pattern, fp32 source with convert-to-bf16 (v_cvt_pk_bf16_f32 pairs).
static __device__ __forceinline__ v16bf frag_rows_f32(const float* __restrict__ p,
                                                      int ld, int rowbase, int colbase, int lane) {
  const int m  = lane & 15;
  const int hi = (lane & 16) ? 8 : 0;
  const float* row = p + (size_t)(rowbase + m) * ld + colbase + hi;
  const v8f lo = *(const v8f*)(row);
  const v8f h8 = *(const v8f*)(row + 16);
  v16bf a;
#pragma unroll
  for (int i = 0; i < 8; ++i) { a[i] = (__bf16)lo[i]; a[i + 8] = (__bf16)h8[i]; }
  return a;
}

// --- kernel 0: fp32 -> bf16 weight conversion WITH transpose ---------------
// Weights are only ever used as B matrices; store Wt[n][k] so B-fragments
// are row-contiguous 16-byte loads.
__global__ void cvt_weights(const float* __restrict__ w0, const float* __restrict__ w1,
                            const float* __restrict__ w2, const float* __restrict__ w3,
                            const float* __restrict__ w4,
                            bf16_t* __restrict__ o0, bf16_t* __restrict__ o1,
                            bf16_t* __restrict__ o2, bf16_t* __restrict__ o3,
                            bf16_t* __restrict__ o4) {
  const int t = blockIdx.x * blockDim.x + threadIdx.x;
  if (t < DMODEL * DMODEL) {
    const int n = t >> 7, k = t & 127;
    const int src = k * DMODEL + n;
    o0[t] = (__bf16)w0[src]; o1[t] = (__bf16)w1[src]; o2[t] = (__bf16)w2[src];
    o3[t] = (__bf16)w3[src]; o4[t] = (__bf16)w4[src];
  }
}

// --- kernel 1: residual + Q/K/V projections (WMMA) -------------------------
// One WG per 16 token rows; wave w owns output-column tile w (16 cols) for
// all four GEMMs. Q/K stored row-major bf16; V stored TRANSPOSED per head:
// Vt[(b*128 + h*32 + dv) * SEQ + s]  -> contiguous keys for the P*V B-frags.
__global__ void __launch_bounds__(256)
proj_kernel(const float* __restrict__ inQ, const float* __restrict__ inK,
            const float* __restrict__ inV,
            const bf16_t* __restrict__ Wfc0t, const bf16_t* __restrict__ WQt,
            const bf16_t* __restrict__ WKt,   const bf16_t* __restrict__ WVt,
            float* __restrict__ resid, bf16_t* __restrict__ Qb,
            bf16_t* __restrict__ Kb,   bf16_t* __restrict__ Vt) {
  const int rowbase = blockIdx.x * 16;
  const int lane = threadIdx.x & 31;
  const int wave = threadIdx.x >> 5;           // column tile 0..7
  v8f accR = {}, accQ = {}, accK = {}, accV = {};
#pragma unroll
  for (int kk = 0; kk < 4; ++kk) {
    const int kb = kk * 32;
    v16bf aQ = frag_rows_f32(inQ, DMODEL, rowbase, kb, lane);
    v16bf aK = frag_rows_f32(inK, DMODEL, rowbase, kb, lane);
    v16bf aV = frag_rows_f32(inV, DMODEL, rowbase, kb, lane);
    v16bf b0 = frag_rows_bf16(Wfc0t, DMODEL, wave * 16, kb, lane);
    v16bf b1 = frag_rows_bf16(WQt,   DMODEL, wave * 16, kb, lane);
    v16bf b2 = frag_rows_bf16(WKt,   DMODEL, wave * 16, kb, lane);
    v16bf b3 = frag_rows_bf16(WVt,   DMODEL, wave * 16, kb, lane);
    accR = WMMA_BF16(aQ, b0, accR);
    accQ = WMMA_BF16(aQ, b1, accQ);
    accK = WMMA_BF16(aK, b2, accK);
    accV = WMMA_BF16(aV, b3, accV);
  }
  // C/D layout: lane -> N = lane%16, vgpr r -> M = r + (lane>=16 ? 8 : 0)
  const int n  = wave * 16 + (lane & 15);      // output feature 0..127
  const int mh = (lane & 16) ? 8 : 0;
  const int bb = rowbase >> 11;                // batch
  const int sb = rowbase & (SEQ - 1);          // seq position base
#pragma unroll
  for (int r = 0; r < 8; ++r) {
    const size_t idx = (size_t)(rowbase + r + mh) * DMODEL + n;
    resid[idx] = accR[r];
    Qb[idx] = (__bf16)accQ[r];
    Kb[idx] = (__bf16)accK[r];
    // transposed V: row = b*128 + n (== (b*NH + h)*32 + dv), col = seq pos
    Vt[((size_t)bb * DMODEL + n) * SEQ + sb + r + mh] = (__bf16)accV[r];
  }
}

// --- kernel 2: attention for one (b, h, 16-query tile) ---------------------
// LDS: scores[16][2048] fp32 (128 KB) + reduction scratch + ctx accumulator.
__global__ void __launch_bounds__(256)
attn_kernel(const bf16_t* __restrict__ Qb, const bf16_t* __restrict__ Kb,
            const bf16_t* __restrict__ Vt, const unsigned char* __restrict__ mask,
            float* __restrict__ attn_out, bf16_t* __restrict__ ctx_out) {
  extern __shared__ float smem[];
  float* sc     = smem;                 // 16*2048
  float* red    = sc + 16 * SEQ;        // 256
  float* rowmax = red + 256;            // 16
  float* rowsum = rowmax + 16;          // 16
  float* ctx    = rowsum + 16;          // 16*32 = 512

  const int wg   = blockIdx.x;
  const int qt   = wg & (SEQ / 16 - 1);       // 0..127
  const int h    = (wg >> 7) & (NH - 1);
  const int b    = wg >> 9;
  const int lane = threadIdx.x & 31;
  const int wave = threadIdx.x >> 5;
  const int rowg = b * SEQ + qt * 16;         // query row base in [B*S]
  const int col0 = h * DK;
  const float SCALE = 0.17677669529663687f;   // 1/sqrt(32)

  // Q fragment (16 queries x 32 dk), shared by all waves
  const v16bf aq = frag_rows_bf16(Qb, DMODEL, rowg, col0, lane);

  // ---- scores = Q K^T : wave w handles key tiles j = w, w+8, ... (16 tiles)
  for (int jj = 0; jj < 16; ++jj) {
    const int j = wave + jj * 8;
    // prefetch next K tile for this wave (global_prefetch_b8)
    if (jj < 15) {
      const bf16_t* nxt = Kb + (size_t)(b * SEQ + (j + 8) * 16 + (lane & 15)) * DMODEL + col0;
      __builtin_prefetch(nxt, 0, 3);
    }
    const v16bf bk = frag_rows_bf16(Kb, DMODEL, b * SEQ + j * 16, col0, lane);
    v8f c = {};
    c = WMMA_BF16(aq, bk, c);
    const int n   = lane & 15;
    const int key = j * 16 + n;
    const int mh  = (lane & 16) ? 8 : 0;
#pragma unroll
    for (int r = 0; r < 8; ++r) {
      sc[(r + mh) * SEQ + key] = c[r] * SCALE;   // mask applied in max-pass
    }
  }
  __syncthreads();

  // ---- softmax: 16 threads per row, 128 contiguous cols each (float4 ops).
  // Apply the attention mask here (coalesced uint reads) and find row max.
  const int row = threadIdx.x >> 4;
  const int seg = threadIdx.x & 15;
  const int q   = qt * 16 + row;
  v4f* srow4 = (v4f*)(sc + row * SEQ + seg * 128);
  {
    float mx = -3.4e38f;
    const unsigned int* m32 =
        (const unsigned int*)(mask + ((size_t)b * SEQ + q) * SEQ + seg * 128);
    for (int i = 0; i < 32; ++i) {
      const unsigned int mm = m32[i];
      v4f s = srow4[i];
#pragma unroll
      for (int jx = 0; jx < 4; ++jx) {
        if ((mm >> (8 * jx)) & 0xffu) s[jx] = -1e9f;
        mx = fmaxf(mx, s[jx]);
      }
      srow4[i] = s;
    }
    red[row * 16 + seg] = mx;
  }
  __syncthreads();
  if (threadIdx.x < 16) {
    float mx = red[threadIdx.x * 16];
    for (int i = 1; i < 16; ++i) mx = fmaxf(mx, red[threadIdx.x * 16 + i]);
    rowmax[threadIdx.x] = mx;
  }
  __syncthreads();
  {
    const float mx = rowmax[row];
    float ssum = 0.f;
    for (int i = 0; i < 32; ++i) {
      v4f x = srow4[i];
#pragma unroll
      for (int jx = 0; jx < 4; ++jx) { x[jx] = __expf(x[jx] - mx); ssum += x[jx]; }
      srow4[i] = x;
    }
    red[row * 16 + seg] = ssum;
  }
  __syncthreads();
  if (threadIdx.x < 16) {
    float s = 0.f;
    for (int i = 0; i < 16; ++i) s += red[threadIdx.x * 16 + i];
    rowsum[threadIdx.x] = s;
  }
  __syncthreads();
  {
    const float inv = 1.0f / rowsum[row];
    float* arow = attn_out + ((size_t)(b * NH + h) * SEQ + q) * SEQ + seg * 128;
    for (int i = 0; i < 32; ++i) {
      const v4f p = srow4[i] * inv;
      srow4[i] = p;                                        // keep probs for P·V
      __builtin_nontemporal_store(p, (v4f*)(arow) + i);    // b128 NT stream
    }
  }
  // zero ctx accumulator (512 floats) before cross-wave reduction
  ctx[threadIdx.x]       = 0.f;
  ctx[threadIdx.x + 256] = 0.f;
  __syncthreads();

  // ---- context = P · V : wave w covers keys [w*256, w*256+256), 2 N-tiles.
  // B-fragments come from transposed Vt -> contiguous 16-byte key runs.
  v8f acc0 = {}, acc1 = {};
  const int mA  = lane & 15;
  const int hiA = (lane & 16) ? 8 : 0;
  const int nB  = lane & 15;
  const bf16_t* vbase = Vt + ((size_t)b * DMODEL + col0 + nB) * SEQ;
  for (int kt = 0; kt < 8; ++kt) {
    const int keybase = wave * 256 + kt * 32;
    v16bf ap;
    {
      const float* prow = sc + mA * SEQ + keybase + hiA;
      const v8f lo = *(const v8f*)(prow);
      const v8f h8 = *(const v8f*)(prow + 16);
#pragma unroll
      for (int i = 0; i < 8; ++i) { ap[i] = (__bf16)lo[i]; ap[i + 8] = (__bf16)h8[i]; }
    }
    const bf16_t* v0p = vbase + keybase + hiA;
    const bf16_t* v1p = v0p + (size_t)16 * SEQ;
    const v16bf bv0 = combine8(*(const v8bf*)(v0p), *(const v8bf*)(v0p + 16));
    const v16bf bv1 = combine8(*(const v8bf*)(v1p), *(const v8bf*)(v1p + 16));
    acc0 = WMMA_BF16(ap, bv0, acc0);
    acc1 = WMMA_BF16(ap, bv1, acc1);
  }
  {
    const int mh = (lane & 16) ? 8 : 0;
#pragma unroll
    for (int r = 0; r < 8; ++r) {
      const int m = r + mh;
      atomicAdd(&ctx[m * 32 + nB],      acc0[r]);
      atomicAdd(&ctx[m * 32 + 16 + nB], acc1[r]);
    }
  }
  __syncthreads();

  // write context (bf16) in [B, S, H*DV] layout
  for (int t = threadIdx.x; t < 512; t += 256) {
    const int m = t >> 5;
    const int n = t & 31;
    ctx_out[(size_t)(b * SEQ + qt * 16 + m) * DMODEL + col0 + n] = (__bf16)ctx[t];
  }
}

// --- kernel 3: out = LN(context @ W_fc + residual) -------------------------
__global__ void __launch_bounds__(256)
out_ln_kernel(const bf16_t* __restrict__ Cb, const bf16_t* __restrict__ Wfct,
              const float* __restrict__ resid, const float* __restrict__ gamma,
              const float* __restrict__ beta, float* __restrict__ out) {
  __shared__ float o[16 * DMODEL];
  __shared__ float red1[256], red2[256], mu[16], rs[16];
  const int rowbase = blockIdx.x * 16;
  const int lane = threadIdx.x & 31;
  const int wave = threadIdx.x >> 5;

  v8f acc = {};
#pragma unroll
  for (int kk = 0; kk < 4; ++kk) {
    v16bf a = frag_rows_bf16(Cb,   DMODEL, rowbase,  kk * 32, lane);
    v16bf w = frag_rows_bf16(Wfct, DMODEL, wave * 16, kk * 32, lane);
    acc = WMMA_BF16(a, w, acc);
  }
  {
    const int n  = wave * 16 + (lane & 15);
    const int mh = (lane & 16) ? 8 : 0;
#pragma unroll
    for (int r = 0; r < 8; ++r) {
      const int m = r + mh;
      o[m * DMODEL + n] = acc[r] + resid[(size_t)(rowbase + m) * DMODEL + n];
    }
  }
  __syncthreads();

  const int row = threadIdx.x >> 4;
  const int seg = threadIdx.x & 15;
  {
    float s1 = 0.f, s2 = 0.f;
    const float* orow = o + row * DMODEL + seg * 8;
#pragma unroll
    for (int i = 0; i < 8; ++i) { const float x = orow[i]; s1 += x; s2 += x * x; }
    red1[row * 16 + seg] = s1;
    red2[row * 16 + seg] = s2;
  }
  __syncthreads();
  if (threadIdx.x < 16) {
    float s1 = 0.f, s2 = 0.f;
    for (int i = 0; i < 16; ++i) { s1 += red1[threadIdx.x * 16 + i]; s2 += red2[threadIdx.x * 16 + i]; }
    const float mean = s1 * (1.0f / DMODEL);
    const float var  = s2 * (1.0f / DMODEL) - mean * mean;
    mu[threadIdx.x] = mean;
    rs[threadIdx.x] = rsqrtf(var + LN_EPS);
  }
  __syncthreads();
  {
    const float m = mu[row], r = rs[row];
    v4f o0, o1;
#pragma unroll
    for (int i = 0; i < 4; ++i) {
      const int c0 = seg * 8 + i, c1 = seg * 8 + 4 + i;
      o0[i] = (o[row * DMODEL + c0] - m) * r * gamma[c0] + beta[c0];
      o1[i] = (o[row * DMODEL + c1] - m) * r * gamma[c1] + beta[c1];
    }
    v4f* dst = (v4f*)(out + (size_t)(rowbase + row) * DMODEL + seg * 8);
    dst[0] = o0;
    dst[1] = o1;
  }
}

// ---------------------------------------------------------------------------
extern "C" void kernel_launch(void* const* d_in, const int* in_sizes, int n_in,
                              void* d_out, int out_size, void* d_ws, size_t ws_size,
                              hipStream_t stream) {
  (void)in_sizes; (void)n_in; (void)out_size; (void)ws_size;
  const float* inQ   = (const float*)d_in[0];
  const float* inK   = (const float*)d_in[1];
  const float* inV   = (const float*)d_in[2];
  const unsigned char* msk = (const unsigned char*)d_in[3];  // jax bool -> 1 byte
  const float* Wfc0  = (const float*)d_in[4];
  const float* WQ    = (const float*)d_in[5];
  const float* WK    = (const float*)d_in[6];
  const float* WV    = (const float*)d_in[7];
  const float* Wfc   = (const float*)d_in[8];
  const float* gamma = (const float*)d_in[9];
  const float* beta  = (const float*)d_in[10];

  float* out  = (float*)d_out;                         // [B,S,D]
  float* attn = out + (size_t)ROWS * DMODEL;           // [B,H,S,S]

  // workspace carve-up (~24.3 MB total)
  char* ws = (char*)d_ws;
  size_t off = 0;
  auto take = [&](size_t nbytes) -> char* {
    char* p = ws + off;
    off += (nbytes + 255) & ~(size_t)255;
    return p;
  };
  bf16_t* Wfc0t = (bf16_t*)take(DMODEL * DMODEL * sizeof(bf16_t));
  bf16_t* WQt   = (bf16_t*)take(DMODEL * DMODEL * sizeof(bf16_t));
  bf16_t* WKt   = (bf16_t*)take(DMODEL * DMODEL * sizeof(bf16_t));
  bf16_t* WVt   = (bf16_t*)take(DMODEL * DMODEL * sizeof(bf16_t));
  bf16_t* Wfct  = (bf16_t*)take(DMODEL * DMODEL * sizeof(bf16_t));
  bf16_t* Qb    = (bf16_t*)take((size_t)ROWS * DMODEL * sizeof(bf16_t));
  bf16_t* Kb    = (bf16_t*)take((size_t)ROWS * DMODEL * sizeof(bf16_t));
  bf16_t* Vt    = (bf16_t*)take((size_t)ROWS * DMODEL * sizeof(bf16_t));
  float*  resid = (float*)take((size_t)ROWS * DMODEL * sizeof(float));
  bf16_t* Cb    = (bf16_t*)take((size_t)ROWS * DMODEL * sizeof(bf16_t));

  cvt_weights<<<dim3((DMODEL * DMODEL + 255) / 256), dim3(256), 0, stream>>>(
      Wfc0, WQ, WK, WV, Wfc, Wfc0t, WQt, WKt, WVt, Wfct);

  proj_kernel<<<dim3(ROWS / 16), dim3(256), 0, stream>>>(
      inQ, inK, inV, Wfc0t, WQt, WKt, WVt, resid, Qb, Kb, Vt);

  const size_t smem = (size_t)(16 * SEQ + 256 + 16 + 16 + 512) * sizeof(float); // ~131 KB
  attn_kernel<<<dim3(BATCH * NH * (SEQ / 16)), dim3(256), smem, stream>>>(
      Qb, Kb, Vt, msk, attn, Cb);

  out_ln_kernel<<<dim3(ROWS / 16), dim3(256), 0, stream>>>(
      Cb, Wfct, resid, gamma, beta, out);
}
